// LSTMModel_32762010534386
// MI455X (gfx1250) — matrix operand order repaired and checked by reading, not compiled, not run on hardware
//
#include <hip/hip_runtime.h>
#include <hip/hip_bf16.h>
#include <math.h>

#define B_   512
#define T_   256
#define DIN_ 128   // input width of both layers (I=128, 2H=128)
#define H_   64
#define G_   256   // 4*H gates

typedef __attribute__((ext_vector_type(16))) __bf16 v16bf;
typedef __attribute__((ext_vector_type(8)))  __bf16 v8bf;
typedef __attribute__((ext_vector_type(8)))  float  v8f;

__device__ __forceinline__ v16bf pack16(v8bf a, v8bf b) {
    v16bf r;
#pragma unroll
    for (int i = 0; i < 8; ++i) { r[i] = a[i]; r[8 + i] = b[i]; }
    return r;
}

// A fragment (16xK tile, rows striped per lane): rows m0..m0+15 via lane&15,
// cols k0..k0+31.  Lanes 0-15: K {0..7,16..23}; lanes 16-31: K {8..15,24..31}.
__device__ __forceinline__ v16bf load_a(const __bf16* base, int row_stride,
                                        int k0, int lane) {
    const int half = lane >> 4, m = lane & 15;
    const __bf16* p = base + (size_t)m * row_stride + k0 + half * 8;
    v8bf c0 = *(const v8bf*)p;
    v8bf c1 = *(const v8bf*)(p + 16);
    return pack16(c0, c1);
}

// B fragment from W (N x K row-major): columns n0..n0+15 (lane&15).
// Lanes 0-15: K {0..7,16..23}; lanes 16-31: K {8..15,24..31}.
__device__ __forceinline__ v16bf load_b(const __bf16* w, int ldk, int n0,
                                        int k0, int lane) {
    const int half = lane >> 4, n = lane & 15;
    const __bf16* p = w + (size_t)(n0 + n) * ldk + k0 + half * 8;
    v8bf c0 = *(const v8bf*)p;
    v8bf c1 = *(const v8bf*)(p + 16);
    return pack16(c0, c1);
}

// Branchless activations on v_exp_f32 (no EXEC-mask special-casing).
__device__ __forceinline__ float sigm(float x) {
    return 1.0f / (1.0f + __expf(-x));
}
__device__ __forceinline__ float tanh_fast(float x) {
    // tanh(x) = 1 - 2/(exp(2x)+1); saturates correctly for |x| large.
    return 1.0f - 2.0f / (__expf(2.0f * x) + 1.0f);
}

// One bidirectional-LSTM layer. grid = (B/16, 2 directions), block = 128 (4 waves).
// Each workgroup owns a 16-row batch tile and walks all T steps locally:
//   gates(16x256) = x_t(16x128) @ w_ih^T + h(16x64) @ w_hh^T + b   (WMMA bf16)
// Wave w owns hidden columns [16w,16w+16) -> gate tiles at N = 64q + 16w.
// h is ping-pong buffered in LDS so only one barrier is needed per timestep.
__global__ __launch_bounds__(128)
void lstm_layer_kernel(const __bf16* __restrict__ xin,  // (B,T,128) bf16
                       __bf16* __restrict__ hout,       // seq (B,T,128) or last (B,128)
                       const __bf16* __restrict__ wih_f, const __bf16* __restrict__ whh_f,
                       const float*  __restrict__ b_f,
                       const __bf16* __restrict__ wih_r, const __bf16* __restrict__ whh_r,
                       const float*  __restrict__ b_r,
                       int store_all) {
    __shared__ __bf16 h_lds[2][16 * 64];

    const int dir  = blockIdx.y;
    const int b0   = blockIdx.x * 16;
    const int tid  = threadIdx.x;
    const int lane = tid & 31;
    const int wv   = tid >> 5;
    const int nb   = wv * 16;            // wave's hidden-column base
    const int nl   = lane & 15;          // column within tile
    const int mrow = (lane >> 4) * 8;    // row base for D/C fragment

    const __bf16* wih = dir ? wih_r : wih_f;
    const __bf16* whh = dir ? whh_r : whh_f;
    const float*  bia = dir ? b_r   : b_f;
    const int colbase = dir ? 64 : 0;

    // h0 = 0 in buffer 0
    for (int i = tid; i < 16 * 64; i += 128) h_lds[0][i] = (__bf16)0.0f;
    __syncthreads();

    float bq[4];
#pragma unroll
    for (int q = 0; q < 4; ++q) bq[q] = bia[64 * q + nb + nl];

    v8f cst;  // cell state for this wave's 16 columns x 8 rows
#pragma unroll
    for (int r = 0; r < 8; ++r) cst[r] = 0.0f;

    for (int s = 0; s < T_; ++s) {
        const int t = dir ? (T_ - 1 - s) : s;
        const __bf16* hcur = h_lds[s & 1];
        __bf16*       hnxt = h_lds[(s + 1) & 1];

        v8f acc[4];
#pragma unroll
        for (int q = 0; q < 4; ++q)
#pragma unroll
            for (int r = 0; r < 8; ++r) acc[q][r] = bq[q];

        // ---- fetch all A fragments for this step up front ----
        const __bf16* xbase = xin + ((size_t)b0 * T_ + t) * DIN_;
        v16bf af[4];
#pragma unroll
        for (int kk = 0; kk < 4; ++kk)
            af[kk] = load_a(xbase, T_ * DIN_, kk * 32, lane);
        v16bf ar0 = load_a(hcur, 64, 0,  lane);
        v16bf ar1 = load_a(hcur, 64, 32, lane);

        // ---- input GEMM: x_t(16x128) @ w_ih^T ----
#pragma unroll
        for (int kk = 0; kk < 4; ++kk) {
#pragma unroll
            for (int q = 0; q < 4; ++q) {
                v16bf bb = load_b(wih, DIN_, 64 * q + nb, kk * 32, lane);
                acc[q] = __builtin_amdgcn_wmma_f32_16x16x32_bf16(
                    false, af[kk], false, bb, (short)0, acc[q], false, false);
            }
        }

        // ---- recurrent GEMM: h(16x64) @ w_hh^T ----
#pragma unroll
        for (int q = 0; q < 4; ++q) {
            v16bf bb0 = load_b(whh, H_, 64 * q + nb, 0, lane);
            acc[q] = __builtin_amdgcn_wmma_f32_16x16x32_bf16(
                false, ar0, false, bb0, (short)0, acc[q], false, false);
            v16bf bb1 = load_b(whh, H_, 64 * q + nb, 32, lane);
            acc[q] = __builtin_amdgcn_wmma_f32_16x16x32_bf16(
                false, ar1, false, bb1, (short)0, acc[q], false, false);
        }

        // ---- elementwise gate math, state update ----
        __bf16 hv[8];
#pragma unroll
        for (int r = 0; r < 8; ++r) {
            float ig = sigm(acc[0][r]);
            float fg = sigm(acc[1][r]);
            float gg = tanh_fast(acc[2][r]);
            float og = sigm(acc[3][r]);
            float cc = fg * cst[r] + ig * gg;
            cst[r] = cc;
            hv[r] = (__bf16)(og * tanh_fast(cc));
        }

        // ---- publish h: ping-pong LDS (for next step) + global output ----
#pragma unroll
        for (int r = 0; r < 8; ++r) {
            const int m = mrow + r;
            hnxt[m * 64 + nb + nl] = hv[r];
            if (store_all) {
                hout[((size_t)(b0 + m) * T_ + t) * 128 + colbase + nb + nl] = hv[r];
            } else if (t == T_ - 1) {
                hout[(size_t)(b0 + m) * 128 + colbase + nb + nl] = hv[r];
            }
        }
        __syncthreads();  // writes to hnxt visible before next step's reads
    }
}

__global__ void f32_to_bf16_kernel(const float* __restrict__ in,
                                   __bf16* __restrict__ out, int n) {
    int i = blockIdx.x * blockDim.x + threadIdx.x;
    if (i < n) out[i] = (__bf16)in[i];
}

// out(512,3) = last(512,128) @ fc_w^T + fc_b ; `last` = layer-1 t=T-1 slice
__global__ void fc_kernel(const __bf16* __restrict__ h1last,
                          const float* __restrict__ fcw,
                          const float* __restrict__ fcb,
                          float* __restrict__ out) {
    int b = blockIdx.x * blockDim.x + threadIdx.x;
    if (b >= B_) return;
    const __bf16* hp = h1last + (size_t)b * 128;
#pragma unroll
    for (int c = 0; c < 3; ++c) {
        float s = fcb[c];
        for (int j = 0; j < 128; ++j) s += (float)hp[j] * fcw[c * 128 + j];
        out[b * 3 + c] = s;
    }
}

extern "C" void kernel_launch(void* const* d_in, const int* in_sizes, int n_in,
                              void* d_out, int out_size, void* d_ws, size_t ws_size,
                              hipStream_t stream) {
    const float* x = (const float*)d_in[0];
    // per-direction weights: (w_ih, w_hh, b) x {l0f,l0r,l1f,l1r}
    const float* w_ih[4] = {(const float*)d_in[1], (const float*)d_in[4],
                            (const float*)d_in[7], (const float*)d_in[10]};
    const float* w_hh[4] = {(const float*)d_in[2], (const float*)d_in[5],
                            (const float*)d_in[8], (const float*)d_in[11]};
    const float* bias[4] = {(const float*)d_in[3], (const float*)d_in[6],
                            (const float*)d_in[9], (const float*)d_in[12]};
    const float* fc_w = (const float*)d_in[13];
    const float* fc_b = (const float*)d_in[14];

    // ---- workspace layout (bf16 buffers) ----
    const size_t XE = (size_t)B_ * T_ * DIN_;           // 16,777,216 elems
    char* ws = (char*)d_ws;
    __bf16* x_bf = (__bf16*)ws;                          // (B,T,128)
    __bf16* h0   = (__bf16*)(ws + XE * 2);               // (B,T,128) layer-0 seq
    __bf16* h1l  = (__bf16*)(ws + XE * 4);               // (B,128) layer-1 last
    __bf16* wbuf = (__bf16*)(ws + XE * 4 + (size_t)B_ * 128 * 2 + 256);
    __bf16* wih_bf[4];
    __bf16* whh_bf[4];
    {
        __bf16* p = wbuf;
        for (int d = 0; d < 4; ++d) {
            wih_bf[d] = p; p += G_ * DIN_;   // 256x128
            whh_bf[d] = p; p += G_ * H_;     // 256x64
        }
    }

    // ---- prep: convert x + weights to bf16 ----
    {
        int n = (int)XE;
        f32_to_bf16_kernel<<<(n + 255) / 256, 256, 0, stream>>>(x, x_bf, n);
        for (int d = 0; d < 4; ++d) {
            f32_to_bf16_kernel<<<(G_ * DIN_ + 255) / 256, 256, 0, stream>>>(
                w_ih[d], wih_bf[d], G_ * DIN_);
            f32_to_bf16_kernel<<<(G_ * H_ + 255) / 256, 256, 0, stream>>>(
                w_hh[d], whh_bf[d], G_ * H_);
        }
    }

    dim3 grid(B_ / 16, 2), block(128);
    // layer 0: x_bf -> h0 (full sequence, f->cols 0:64, r->cols 64:128)
    lstm_layer_kernel<<<grid, block, 0, stream>>>(
        x_bf, h0, wih_bf[0], whh_bf[0], bias[0], wih_bf[1], whh_bf[1], bias[1], 1);
    // layer 1: h0 -> h1 last slice only
    lstm_layer_kernel<<<grid, block, 0, stream>>>(
        h0, h1l, wih_bf[2], whh_bf[2], bias[2], wih_bf[3], whh_bf[3], bias[3], 0);
    // FC head
    fc_kernel<<<(B_ + 255) / 256, 256, 0, stream>>>(h1l, fc_w, fc_b, (float*)d_out);
}
